// DiffMLAAttention_64922725647104
// MI455X (gfx1250) — matrix operand order repaired and checked by reading, not compiled, not run on hardware
//
#include <hip/hip_runtime.h>
#include <hip/hip_bf16.h>
#include <math.h>

// ---------------------------------------------------------------------------
// Problem constants: D=2048 NH=16 DH=128 DHR=64 DC=1024 B=2 L=2048
// ---------------------------------------------------------------------------
static constexpr int Dd   = 2048;
static constexpr int NHh  = 16;
static constexpr int DHh  = 128;
static constexpr int DHRr = 64;
static constexpr int DCc  = 1024;
static constexpr int Bb   = 2;
static constexpr int Ll   = 2048;
static constexpr int BL   = Bb * Ll;      // 4096 token rows
static constexpr int DQ   = DHh + DHRr;   // 192

typedef __attribute__((ext_vector_type(16))) __bf16 v16bf;
typedef __attribute__((ext_vector_type(8)))  float  v8f;
typedef __attribute__((__vector_size__(4 * sizeof(int)))) int vi4;

__device__ __forceinline__ __bf16 f2bf(float f) { return (__bf16)f; }

// ---------------------------------------------------------------------------
// Async global->LDS staging (CDNA5 GLOBAL_LOAD_ASYNC_TO_LDS_B128, ASYNCcnt).
// Builtin expects: (v4i addrspace(1)*, v4i addrspace(3)*, imm offset, imm cpol)
// Guarded: falls back to a plain 16-byte copy if the builtin is unavailable.
// ---------------------------------------------------------------------------
#if __has_builtin(__builtin_amdgcn_global_load_async_to_lds_b128)
#define HAS_ASYNC_LDS 1
#else
#define HAS_ASYNC_LDS 0
#endif

__device__ __forceinline__ void async_copy16(const void* g, void* l) {
#if HAS_ASYNC_LDS
  __builtin_amdgcn_global_load_async_to_lds_b128(
      (__attribute__((address_space(1))) vi4*)g,
      (__attribute__((address_space(3))) vi4*)l, 0, 0);
#else
  *(uint4*)l = *(const uint4*)g;
#endif
}

__device__ __forceinline__ void wait_async() {
#if HAS_ASYNC_LDS
#if __has_builtin(__builtin_amdgcn_s_wait_asynccnt)
  __builtin_amdgcn_s_wait_asynccnt(0);
#else
  asm volatile("s_wait_asynccnt 0x0" ::: "memory");
#endif
#endif
}

// ---------------------------------------------------------------------------
// WMMA fragment loaders (16x16x32 bf16, wave32; contiguous k = 16h + e, not the ISA table k order).
// All loads below are per-lane CONTIGUOUS (lower to ds_load_b128 pairs).
// ---------------------------------------------------------------------------
// A matrix 16x32 (MxK), row-major [m][k]: lane half h, row = lane&15,
// element e -> K = e + 16h  => 16 contiguous elements per lane.
__device__ __forceinline__ v16bf load_frag_a(const __bf16* p, int row_base, int ld,
                                             int k_base, int lane) {
  int h = (lane >> 4) & 1;
  const __bf16* r = p + (size_t)(row_base + (lane & 15)) * ld + k_base;
  v16bf a;
#pragma unroll
  for (int e = 0; e < 16; ++e) {
    int k = e + 16 * h;
    a[e] = r[k];
  }
  return a;
}

// B matrix 32x16 (KxN) where the source is stored transposed [n][k]
// (weightsT, keys, V-transposed): lane half h, col = lane&15,
// element e -> K = e + 16h  => 16 contiguous elements per lane.
__device__ __forceinline__ v16bf load_frag_bT(const __bf16* p, int col_base, int ld,
                                              int k_base, int lane) {
  int h = (lane >> 4) & 1;
  const __bf16* r = p + (size_t)(col_base + (lane & 15)) * ld + k_base + 16 * h;
  v16bf b;
#pragma unroll
  for (int e = 0; e < 16; ++e) b[e] = r[e];
  return b;
}

__device__ __forceinline__ v8f wmma_bf16(v16bf a, v16bf b, v8f c) {
  return __builtin_amdgcn_wmma_f32_16x16x32_bf16(false, a, false, b, (short)0, c,
                                                 false, false);
}

// ---------------------------------------------------------------------------
// fp32 -> bf16 convert (grid stride)
// ---------------------------------------------------------------------------
__global__ void cvt_f32_bf16_kernel(const float* __restrict__ s,
                                    __bf16* __restrict__ d, size_t n) {
  size_t i = (size_t)blockIdx.x * blockDim.x + threadIdx.x;
  size_t stride = (size_t)gridDim.x * blockDim.x;
  for (; i < n; i += stride) d[i] = f2bf(s[i]);
}

// fp32 [K][N] -> bf16 transposed [N][K]
__global__ void cvt_f32_bf16_t_kernel(const float* __restrict__ s,
                                      __bf16* __restrict__ d, int K, int N) {
  size_t i = (size_t)blockIdx.x * blockDim.x + threadIdx.x;
  size_t total = (size_t)K * N;
  size_t stride = (size_t)gridDim.x * blockDim.x;
  for (; i < total; i += stride) {
    int k = (int)(i % K);
    int n = (int)(i / K);
    d[i] = f2bf(s[(size_t)k * N + n]);
  }
}

// ---------------------------------------------------------------------------
// bf16 GEMM: C(f32)[M,N] = A(bf16)[M,K] @ Bt(bf16)[N,K]^T
// 128x64 block tile, BK=32, 256 threads (8 waves), wave -> 32x32 subtile.
// Double-buffered LDS, async global->LDS staging, one barrier per K-step.
// Requires M%128==0, N%64==0, K%32==0.
// ---------------------------------------------------------------------------
__global__ __launch_bounds__(256) void gemm_bf16_kernel(
    const __bf16* __restrict__ A, const __bf16* __restrict__ Bt,
    float* __restrict__ C, int M, int N, int K) {
  __shared__ __bf16 As[2][128 * 32];  // [m][k]
  __shared__ __bf16 Bs[2][64 * 32];   // [n][k]

  int tid  = threadIdx.x;
  int lane = tid & 31;
  int wave = tid >> 5;          // 0..7
  int wm = (wave >> 1) * 32;    // 0,32,64,96
  int wn = (wave & 1) * 32;     // 0,32
  int bm = blockIdx.y * 128;
  int bn = blockIdx.x * 64;

  v8f acc[2][2] = {};
  int nk = K >> 5;

  // stage tile 0
  {
#pragma unroll
    for (int v = 0; v < 2; ++v) {  // A: 512 chunks / 256 thr
      int idx = (tid + v * 256) * 8;
      int r = idx >> 5, c = idx & 31;
      async_copy16(&A[(size_t)(bm + r) * K + c], &As[0][r * 32 + c]);
    }
    {  // B: 256 chunks / 256 thr
      int idx = tid * 8;
      int n = idx >> 5, c = idx & 31;
      async_copy16(&Bt[(size_t)(bn + n) * K + c], &Bs[0][n * 32 + c]);
    }
  }

  for (int t = 0; t < nk; ++t) {
    wait_async();
    __syncthreads();  // stage t ready for everyone

    int cur = t & 1;
    if (t + 1 < nk) {
      int k0 = (t + 1) * 32;
      int nxt = cur ^ 1;
#pragma unroll
      for (int v = 0; v < 2; ++v) {
        int idx = (tid + v * 256) * 8;
        int r = idx >> 5, c = idx & 31;
        async_copy16(&A[(size_t)(bm + r) * K + k0 + c], &As[nxt][r * 32 + c]);
      }
      {
        int idx = tid * 8;
        int n = idx >> 5, c = idx & 31;
        async_copy16(&Bt[(size_t)(bn + n) * K + k0 + c], &Bs[nxt][n * 32 + c]);
      }
      if (t + 2 < nk) {
        int k2 = (t + 2) * 32;
        __builtin_prefetch(&A[(size_t)(bm + (tid >> 3)) * K + k2], 0, 1);
        __builtin_prefetch(&Bt[(size_t)(bn + (tid >> 3)) * K + k2], 0, 1);
      }
    }

    v16bf af[2], bfr[2];
#pragma unroll
    for (int mi = 0; mi < 2; ++mi)
      af[mi] = load_frag_a(As[cur], wm + mi * 16, 32, 0, lane);
#pragma unroll
    for (int ni = 0; ni < 2; ++ni)
      bfr[ni] = load_frag_bT(Bs[cur], wn + ni * 16, 32, 0, lane);
#pragma unroll
    for (int mi = 0; mi < 2; ++mi)
#pragma unroll
      for (int ni = 0; ni < 2; ++ni)
        acc[mi][ni] = wmma_bf16(af[mi], bfr[ni], acc[mi][ni]);
  }

  int hh = (lane >> 4) & 1;
  int cn = lane & 15;
#pragma unroll
  for (int mi = 0; mi < 2; ++mi)
#pragma unroll
    for (int ni = 0; ni < 2; ++ni) {
      float* cp = C + (size_t)(bm + wm + mi * 16 + 8 * hh) * N +
                  (bn + wn + ni * 16 + cn);
#pragma unroll
      for (int j = 0; j < 8; ++j) cp[(size_t)j * N] = acc[mi][ni][j];
    }
}

// ---------------------------------------------------------------------------
// RMSNorm: out_bf16[row, c] = in[row,c] * rsqrt(mean(in^2)+eps) * w[c]
// ---------------------------------------------------------------------------
__global__ __launch_bounds__(256) void rms_kernel(const float* __restrict__ in,
                                                  const float* __restrict__ w,
                                                  __bf16* __restrict__ out,
                                                  int ncols) {
  int row = blockIdx.x;
  const float* r = in + (size_t)row * ncols;
  float s = 0.f;
  for (int c = threadIdx.x; c < ncols; c += 256) {
    float v = r[c];
    s += v * v;
  }
  __shared__ float red[256];
  red[threadIdx.x] = s;
  __syncthreads();
  for (int off = 128; off > 0; off >>= 1) {
    if (threadIdx.x < off) red[threadIdx.x] += red[threadIdx.x + off];
    __syncthreads();
  }
  float inv = rsqrtf(red[0] / (float)ncols + 1e-6f);
  for (int c = threadIdx.x; c < ncols; c += 256)
    out[(size_t)row * ncols + c] = f2bf(r[c] * inv * w[c]);
}

// ---------------------------------------------------------------------------
// Pack Q: q_bf16 (B,32,L,192): [0:128] from q_c, [128:192] = rope(q_r)
// ---------------------------------------------------------------------------
__global__ void pack_q_kernel(const float* __restrict__ qc,
                              const float* __restrict__ qr,
                              __bf16* __restrict__ q) {
  int idx = blockIdx.x * blockDim.x + threadIdx.x;  // over B*32*L
  if (idx >= Bb * 32 * Ll) return;
  int l = idx % Ll;
  int bj = idx / Ll;
  int j = bj & 31;
  int b = bj >> 5;
  size_t rowBL = (size_t)b * Ll + l;
  const float* qcp = qc + (rowBL * 32 + j) * 128;
  const float* qrp = qr + (rowBL * 32 + j) * 64;
  __bf16* dst = q + ((size_t)bj * Ll + l) * DQ;
#pragma unroll 8
  for (int d = 0; d < 128; ++d) dst[d] = f2bf(qcp[d]);
  for (int i = 0; i < 32; ++i) {
    float invf = powf(10000.0f, -((float)(2 * i) / 64.0f));
    float fr = (float)l * invf;
    float cs = cosf(fr), sn = sinf(fr);
    float x1 = qrp[i], x2 = qrp[i + 32];
    dst[128 + i]      = f2bf(x1 * cs - x2 * sn);
    dst[128 + 32 + i] = f2bf(x2 * cs + x1 * sn);
  }
}

// ---------------------------------------------------------------------------
// Pack K,V: k_bf16 (B,16,L,192) = [k_c | rope(k_r)]
//           v_bf16 TRANSPOSED (B,16,128,L)   (for contiguous P@V fragments)
// ---------------------------------------------------------------------------
__global__ void pack_kv_kernel(const float* __restrict__ kc,
                               const float* __restrict__ kr,
                               const float* __restrict__ vr,
                               __bf16* __restrict__ kout,
                               __bf16* __restrict__ vout) {
  int idx = blockIdx.x * blockDim.x + threadIdx.x;  // over B*16*L
  if (idx >= Bb * NHh * Ll) return;
  int l = idx % Ll;
  int bh = idx / Ll;
  int h = bh % NHh;
  int b = bh / NHh;
  size_t rowBL = (size_t)b * Ll + l;
  const float* kcp = kc + (rowBL * NHh + h) * 128;
  const float* krp = kr + rowBL * 64;
  const float* vp  = vr + (rowBL * NHh + h) * 128;
  __bf16* kd = kout + ((size_t)bh * Ll + l) * DQ;
  __bf16* vd = vout + (size_t)bh * 128 * Ll + l;  // [bh][d][l]
#pragma unroll 8
  for (int d = 0; d < 128; ++d) {
    kd[d] = f2bf(kcp[d]);
    vd[(size_t)d * Ll] = f2bf(vp[d]);
  }
  for (int i = 0; i < 32; ++i) {
    float invf = powf(10000.0f, -((float)(2 * i) / 64.0f));
    float fr = (float)l * invf;
    float cs = cosf(fr), sn = sinf(fr);
    float x1 = krp[i], x2 = krp[i + 32];
    kd[128 + i]      = f2bf(x1 * cs - x2 * sn);
    kd[128 + 32 + i] = f2bf(x2 * cs + x1 * sn);
  }
}

// ---------------------------------------------------------------------------
// lambda = sigmoid(x @ W_lw + b)
// ---------------------------------------------------------------------------
__global__ void lambda_kernel(const float* __restrict__ x,
                              const float* __restrict__ wlw,
                              const float* __restrict__ wlb,
                              float* __restrict__ lam) {
  int idx = blockIdx.x * blockDim.x + threadIdx.x;  // over B*L*NH
  if (idx >= BL * NHh) return;
  int h = idx % NHh;
  int row = idx / NHh;
  const float* xr = x + (size_t)row * Dd;
  float acc = wlb[h];
  for (int d = 0; d < Dd; ++d) acc += xr[d] * wlw[(size_t)d * NHh + h];
  lam[idx] = 1.0f / (1.0f + expf(-acc));
}

// ---------------------------------------------------------------------------
// Flash attention, WMMA bf16, causal, online softmax.
//   grid = (L/64, B*32), 128 threads (4 waves), 32-key tiles.
// Q (B*32, L, 192), K (B*16, L, 192), Vt (B*16, 128, L), O f32 (B*32, L, 128)
// ---------------------------------------------------------------------------
__global__ __launch_bounds__(128) void attn_kernel(
    const __bf16* __restrict__ Q, const __bf16* __restrict__ Kd,
    const __bf16* __restrict__ Vt, float* __restrict__ O) {
  __shared__ __bf16 Qs[64 * DQ];     // 24 KB [q][dq]
  __shared__ __bf16 Ks[32 * DQ];     // 12 KB [key][dq]
  __shared__ __bf16 Vs[128 * 32];    //  8 KB [dv][key]
  __shared__ __bf16 Ps[4][16 * 32];  //  4 KB per-wave P patch

  int qb = blockIdx.x;
  int bj = blockIdx.y;  // b*32 + j, j = 2h+p
  int b = bj >> 5;
  int j = bj & 31;
  int h = j >> 1;

  int tid = threadIdx.x;
  int lane = tid & 31;
  int wave = tid >> 5;
  int half = (lane >> 4) & 1;
  int ln = lane & 15;

  const __bf16* qp = Q + ((size_t)bj * Ll + qb * 64) * DQ;
  const __bf16* kp = Kd + ((size_t)(b * NHh + h) * Ll) * DQ;
  const __bf16* vp = Vt + (size_t)(b * NHh + h) * 128 * Ll;

  // stage Q tile 64x192 (async): 1536 chunks / 128 threads = 12 each
#pragma unroll
  for (int c = 0; c < 12; ++c) {
    int idx = (tid + c * 128) * 8;
    int r = idx / DQ, col = idx % DQ;
    async_copy16(&qp[(size_t)r * DQ + col], &Qs[r * DQ + col]);
  }

  float m_r[8], l_r[8];
  v8f o[8] = {};
#pragma unroll
  for (int i = 0; i < 8; ++i) { m_r[i] = -3.0e38f; l_r[i] = 0.f; }

  const float scale = rsqrtf((float)DQ);
  int kbmax = 2 * qb + 1;

  for (int kb = 0; kb <= kbmax; ++kb) {
    // stage K tile 32x192 (6 chunks/thr) and Vt tile 128x32 (4 chunks/thr)
#pragma unroll
    for (int c = 0; c < 6; ++c) {
      int idx = (tid + c * 128) * 8;
      int r = idx / DQ, col = idx % DQ;
      async_copy16(&kp[((size_t)kb * 32 + r) * DQ + col], &Ks[r * DQ + col]);
    }
#pragma unroll
    for (int c = 0; c < 4; ++c) {
      int idx = (tid + c * 128) * 8;
      int d = idx >> 5, kk = idx & 31;
      async_copy16(&vp[(size_t)d * Ll + kb * 32 + kk], &Vs[d * 32 + kk]);
    }
    wait_async();
    __syncthreads();

    // S = Q K^T : 2 key subtiles x 6 k-steps over dq=192
    v8f S[2] = {};
#pragma unroll
    for (int kt = 0; kt < 6; ++kt) {
      v16bf a = load_frag_a(Qs, wave * 16, DQ, kt * 32, lane);
#pragma unroll
      for (int sub = 0; sub < 2; ++sub) {
        v16bf bb = load_frag_bT(Ks, sub * 16, DQ, kt * 32, lane);
        S[sub] = wmma_bf16(a, bb, S[sub]);
      }
    }

    // scale + causal mask + row max
    float mloc[8];
#pragma unroll
    for (int i = 0; i < 8; ++i) mloc[i] = -3.0e38f;
#pragma unroll
    for (int sub = 0; sub < 2; ++sub) {
      int kg = kb * 32 + sub * 16 + ln;
#pragma unroll
      for (int i = 0; i < 8; ++i) {
        int qg = qb * 64 + wave * 16 + i + 8 * half;
        float s = S[sub][i] * scale;
        if (kg > qg) s = -3.0e38f;
        S[sub][i] = s;
        mloc[i] = fmaxf(mloc[i], s);
      }
    }
#pragma unroll
    for (int i = 0; i < 8; ++i) {
      float v = mloc[i];
#pragma unroll
      for (int mm = 1; mm < 16; mm <<= 1) v = fmaxf(v, __shfl_xor(v, mm, 32));
      mloc[i] = v;
    }

    float alpha[8];
#pragma unroll
    for (int i = 0; i < 8; ++i) {
      float mnew = fmaxf(m_r[i], mloc[i]);
      alpha[i] = expf(m_r[i] - mnew);
      m_r[i] = mnew;
    }

    float lloc[8];
#pragma unroll
    for (int i = 0; i < 8; ++i) lloc[i] = 0.f;
#pragma unroll
    for (int sub = 0; sub < 2; ++sub)
#pragma unroll
      for (int i = 0; i < 8; ++i) {
        float p = expf(S[sub][i] - m_r[i]);
        S[sub][i] = p;
        lloc[i] += p;
      }
#pragma unroll
    for (int i = 0; i < 8; ++i) {
      float v = lloc[i];
#pragma unroll
      for (int mm = 1; mm < 16; mm <<= 1) v += __shfl_xor(v, mm, 32);
      l_r[i] = l_r[i] * alpha[i] + v;
    }

    // rescale O
#pragma unroll
    for (int n = 0; n < 8; ++n)
#pragma unroll
      for (int i = 0; i < 8; ++i) o[n][i] *= alpha[i];

    // stage P (bf16) in per-wave LDS patch: [row 0..15][key 0..31]
#pragma unroll
    for (int sub = 0; sub < 2; ++sub)
#pragma unroll
      for (int i = 0; i < 8; ++i)
        Ps[wave][(i + 8 * half) * 32 + sub * 16 + ln] = f2bf(S[sub][i]);

    // O += P @ V : single 32-deep k step, 8 dv subtiles (V^T contiguous)
    {
      v16bf a = load_frag_a(Ps[wave], 0, 32, 0, lane);
#pragma unroll
      for (int n = 0; n < 8; ++n) {
        v16bf bb = load_frag_bT(Vs, n * 16, 32, 0, lane);
        o[n] = wmma_bf16(a, bb, o[n]);
      }
    }
    __syncthreads();  // all reads done before next tile overwrites
  }

  // normalize + store
  float* op = O + ((size_t)bj * Ll + qb * 64 + wave * 16) * 128;
#pragma unroll
  for (int i = 0; i < 8; ++i) {
    float inv = 1.0f / l_r[i];
#pragma unroll
    for (int n = 0; n < 8; ++n)
      op[(size_t)(i + 8 * half) * 128 + n * 16 + ln] = o[n][i] * inv;
  }
}

// ---------------------------------------------------------------------------
// combine: out_pre[b,l,h*128+d] = attn1 - lam * attn2   (bf16)
// ---------------------------------------------------------------------------
__global__ void combine_kernel(const float* __restrict__ attn,
                               const float* __restrict__ lam,
                               __bf16* __restrict__ outpre) {
  size_t idx = (size_t)blockIdx.x * blockDim.x + threadIdx.x;  // B*L*2048
  if (idx >= (size_t)BL * Dd) return;
  int col = (int)(idx % Dd);
  int row = (int)(idx / Dd);
  int h = col >> 7;
  int d = col & 127;
  int b = row / Ll;
  int l = row % Ll;
  float a1 = attn[(((size_t)b * 32 + 2 * h)     * Ll + l) * 128 + d];
  float a2 = attn[(((size_t)b * 32 + 2 * h + 1) * Ll + l) * 128 + d];
  float lm = lam[(size_t)row * NHh + h];
  outpre[idx] = f2bf(a1 - lm * a2);
}

// ---------------------------------------------------------------------------
// Launcher
// ---------------------------------------------------------------------------
extern "C" void kernel_launch(void* const* d_in, const int* in_sizes, int n_in,
                              void* d_out, int out_size, void* d_ws,
                              size_t ws_size, hipStream_t stream) {
  (void)in_sizes; (void)n_in; (void)out_size; (void)ws_size;

  const float* x    = (const float*)d_in[0];
  const float* wDKV = (const float*)d_in[1];
  const float* kvw  = (const float*)d_in[2];
  const float* wUK  = (const float*)d_in[3];
  const float* wUV  = (const float*)d_in[4];
  const float* wDQ  = (const float*)d_in[5];
  const float* qw   = (const float*)d_in[6];
  const float* wUQ  = (const float*)d_in[7];
  const float* wQR  = (const float*)d_in[8];
  const float* wKR  = (const float*)d_in[9];
  const float* wLW  = (const float*)d_in[10];
  const float* wLB  = (const float*)d_in[11];
  const float* wOUT = (const float*)d_in[12];
  float* out = (float*)d_out;

  char* base = (char*)d_ws;
  size_t off = 0;
  auto take = [&](size_t bytes) {
    size_t o = off;
    off = (off + bytes + 255) & ~(size_t)255;
    return o;
  };
  const size_t NX = (size_t)BL * Dd;

  __bf16* xbf     = (__bf16*)(base + take(NX * 2));
  __bf16* wDKVb   = (__bf16*)(base + take((size_t)Dd * DCc * 2));     // [N=1024][K=2048]
  __bf16* wUKb    = (__bf16*)(base + take((size_t)DCc * 2048 * 2));   // [2048][1024]
  __bf16* wUVb    = (__bf16*)(base + take((size_t)DCc * 2048 * 2));
  __bf16* wDQb    = (__bf16*)(base + take((size_t)Dd * DCc * 2));
  __bf16* wUQb    = (__bf16*)(base + take((size_t)DCc * 4096 * 2));   // [4096][1024]
  __bf16* wQRb    = (__bf16*)(base + take((size_t)DCc * 2048 * 2));
  __bf16* wKRb    = (__bf16*)(base + take((size_t)Dd * DHRr * 2));    // [64][2048]
  __bf16* wOUTb   = (__bf16*)(base + take((size_t)Dd * Dd * 2));
  float*  ckv_raw = (float*)(base + take((size_t)BL * DCc * 4));
  __bf16* ckv_bf  = (__bf16*)(base + take((size_t)BL * DCc * 2));
  float*  cq_raw  = (float*)(base + take((size_t)BL * DCc * 4));
  __bf16* cq_bf   = (__bf16*)(base + take((size_t)BL * DCc * 2));
  float*  kc_raw  = (float*)(base + take((size_t)BL * 2048 * 4));
  float*  v_raw   = (float*)(base + take((size_t)BL * 2048 * 4));
  float*  qc_raw  = (float*)(base + take((size_t)BL * 4096 * 4));
  float*  qr_raw  = (float*)(base + take((size_t)BL * 2048 * 4));
  float*  kr_raw  = (float*)(base + take((size_t)BL * DHRr * 4));
  __bf16* qbf     = (__bf16*)(base + take((size_t)Bb * 32 * Ll * DQ * 2));
  __bf16* kbf     = (__bf16*)(base + take((size_t)Bb * NHh * Ll * DQ * 2));
  __bf16* vbf     = (__bf16*)(base + take((size_t)Bb * NHh * 128 * Ll * 2));
  float*  attn_f  = (float*)(base + take((size_t)Bb * 32 * Ll * 128 * 4));
  float*  lam     = (float*)(base + take((size_t)BL * NHh * 4));
  __bf16* outpre  = (__bf16*)(base + take((size_t)BL * Dd * 2));

  // activations: plain convert; weights: convert + transpose to [N][K]
  cvt_f32_bf16_kernel<<<4096, 256, 0, stream>>>(x, xbf, NX);
  auto cvtT = [&](const float* s, __bf16* d, int K, int N) {
    cvt_f32_bf16_t_kernel<<<4096, 256, 0, stream>>>(s, d, K, N);
  };
  cvtT(wDKV, wDKVb, Dd, DCc);
  cvtT(wUK,  wUKb,  DCc, 2048);
  cvtT(wUV,  wUVb,  DCc, 2048);
  cvtT(wDQ,  wDQb,  Dd, DCc);
  cvtT(wUQ,  wUQb,  DCc, 4096);
  cvtT(wQR,  wQRb,  DCc, 2048);
  cvtT(wKR,  wKRb,  Dd, DHRr);
  cvtT(wOUT, wOUTb, Dd, Dd);

  auto gemm = [&](const __bf16* A, const __bf16* Bt, float* C, int M, int N,
                  int K) {
    dim3 grid(N / 64, M / 128);
    gemm_bf16_kernel<<<grid, 256, 0, stream>>>(A, Bt, C, M, N, K);
  };

  gemm(xbf, wDKVb, ckv_raw, BL, DCc, Dd);
  rms_kernel<<<BL, 256, 0, stream>>>(ckv_raw, kvw, ckv_bf, DCc);
  gemm(ckv_bf, wUKb, kc_raw, BL, 2048, DCc);
  gemm(ckv_bf, wUVb, v_raw,  BL, 2048, DCc);
  gemm(xbf, wDQb, cq_raw, BL, DCc, Dd);
  rms_kernel<<<BL, 256, 0, stream>>>(cq_raw, qw, cq_bf, DCc);
  gemm(cq_bf, wUQb, qc_raw, BL, 4096, DCc);
  gemm(cq_bf, wQRb, qr_raw, BL, 2048, DCc);
  gemm(xbf,  wKRb, kr_raw, BL, DHRr, Dd);

  pack_q_kernel<<<(Bb * 32 * Ll + 255) / 256, 256, 0, stream>>>(qc_raw, qr_raw,
                                                                qbf);
  pack_kv_kernel<<<(Bb * NHh * Ll + 255) / 256, 256, 0, stream>>>(
      kc_raw, kr_raw, v_raw, kbf, vbf);
  lambda_kernel<<<(BL * NHh + 255) / 256, 256, 0, stream>>>(x, wLW, wLB, lam);

  {
    dim3 grid(Ll / 64, Bb * 32);
    attn_kernel<<<grid, 128, 0, stream>>>(qbf, kbf, vbf, attn_f);
  }

  combine_kernel<<<(int)(((size_t)BL * Dd + 255) / 256), 256, 0, stream>>>(
      attn_f, lam, outpre);
  gemm(outpre, wOUTb, out, BL, Dd, Dd);
}
